// SentenceGP_26362509263197
// MI455X (gfx1250) — matrix operand order repaired and checked by reading, not compiled, not run on hardware
//
#include <hip/hip_runtime.h>
#include <hip/hip_bf16.h>
#include <cstdint>
#include <cstddef>

// Problem dims (fixed by reference)
#define B_  16
#define S_  1024
#define D_  1024
#define M_  1024
#define O_  1024
#define R_  (B_*S_)       // 16384 rows of x
#define NS_ITERS 16       // Newton-Schulz iterations (K_ii_reg ~ 2I -> fast convergence)

typedef __attribute__((ext_vector_type(16))) __bf16 v16bf;
typedef __attribute__((ext_vector_type(8)))  __bf16 v8bf;
typedef __attribute__((ext_vector_type(8)))  float  v8f;
typedef __attribute__((ext_vector_type(2)))  float  v2f;

union ABf { v16bf v; v8bf h[2]; };

__device__ __forceinline__ __bf16 f2bf(float f) {
  unsigned u = __builtin_bit_cast(unsigned, f);
  unsigned r = u + 0x7FFFu + ((u >> 16) & 1u);   // round-to-nearest-even
  unsigned short h = (unsigned short)(r >> 16);
  return __builtin_bit_cast(__bf16, h);
}

// ---------------- small helper kernels ----------------

__global__ void rowsumsq_kernel(const float* __restrict__ X, float* __restrict__ out, int cols) {
  int row = blockIdx.x;
  const float* p = X + (size_t)row * cols;
  float s = 0.f;
  for (int c = threadIdx.x; c < cols; c += blockDim.x) { float v = p[c]; s += v * v; }
  __shared__ float red[256];
  red[threadIdx.x] = s; __syncthreads();
  for (int off = 128; off > 0; off >>= 1) {
    if ((int)threadIdx.x < off) red[threadIdx.x] += red[threadIdx.x + off];
    __syncthreads();
  }
  if (threadIdx.x == 0) out[row] = red[0];
}

__global__ void cvt_bf16_kernel(const float* __restrict__ in, __bf16* __restrict__ out, int n) {
  int i = blockIdx.x * blockDim.x + threadIdx.x;
  if (i < n) out[i] = f2bf(in[i]);
}

__global__ void zero_kernel(float* __restrict__ p, int n) {
  int i = blockIdx.x * blockDim.x + threadIdx.x;
  for (; i < n; i += gridDim.x * blockDim.x) p[i] = 0.f;
}

__global__ void rowabsmax_kernel(const float* __restrict__ A, unsigned* __restrict__ nrmbits) {
  int row = blockIdx.x;
  const float* p = A + (size_t)row * M_;
  float s = 0.f;
  for (int c = threadIdx.x; c < M_; c += 256) s += fabsf(p[c]);
  __shared__ float red[256];
  red[threadIdx.x] = s; __syncthreads();
  for (int off = 128; off > 0; off >>= 1) {
    if ((int)threadIdx.x < off) red[threadIdx.x] += red[threadIdx.x + off];
    __syncthreads();
  }
  if (threadIdx.x == 0) atomicMax(nrmbits, __float_as_uint(red[0]));
}

// X0 = A / (||A||_1 * ||A||_inf)   (A symmetric -> ||A||_1 == ||A||_inf, A^T == A)
__global__ void scaleinit_kernel(const float* __restrict__ A, float* __restrict__ X,
                                 const unsigned* __restrict__ nrmbits, int n) {
  float nr = __uint_as_float(*nrmbits);
  float c = 1.0f / (nr * nr);
  int i = blockIdx.x * blockDim.x + threadIdx.x;
  for (; i < n; i += gridDim.x * blockDim.x) X[i] = A[i] * c;
}

__global__ void varout_kernel(const float* __restrict__ vdot, float* __restrict__ out) {
  size_t total = (size_t)R_ * O_;
  size_t i = (size_t)blockIdx.x * blockDim.x + threadIdx.x;
  size_t stride = (size_t)gridDim.x * blockDim.x;
  for (; i < total; i += stride) {
    int row = (int)(i / O_);
    float v = 1.0f - vdot[row];                 // diag(K_xx) == 1 exactly
    float sp = (v > 20.f) ? v : log1pf(expf(v));
    out[i] = sp;
  }
}

// ---------------- shared epilogue for the bf16 GEMM ----------------
// modes: 0 = RBF epilogue, write f32 + bf16 (K_xi)
//        1 = RBF epilogue + noise*I on diagonal, write f32 (K_ii_reg)
//        2 = write bf16 + fused row-dot with Dotf into vdot (A = K_xi * Kinv)
//        3 = write f32 (mean)
__device__ __forceinline__
void epilogue_bf16(v8f acc, int mbase, int ng, int N, int mode,
                   float* __restrict__ Cf, __bf16* __restrict__ Cbf,
                   const float* __restrict__ rn, const float* __restrict__ cn,
                   float sc, float noise,
                   const float* __restrict__ Dotf, float* __restrict__ vdot,
                   int l15) {
  if (mode <= 1) {
    #pragma unroll
    for (int r = 0; r < 8; ++r) {
      int mg = mbase + r;
      float d2 = rn[mg] + cn[ng] - 2.0f * acc[r];
      d2 = fmaxf(d2, 0.0f);
      float v = expf(sc * d2);
      if (mode == 1 && mg == ng) v += noise;
      Cf[(size_t)mg * N + ng] = v;
      if (mode == 0) Cbf[(size_t)mg * N + ng] = f2bf(v);
    }
  } else if (mode == 2) {
    #pragma unroll
    for (int r = 0; r < 8; ++r) {
      int mg = mbase + r;
      float v = acc[r];
      Cbf[(size_t)mg * N + ng] = f2bf(v);
      float part = v * Dotf[(size_t)mg * N + ng];   // A[mg][ng] * K_xi[mg][ng]
      part += __shfl_xor(part, 1, 16);
      part += __shfl_xor(part, 2, 16);
      part += __shfl_xor(part, 4, 16);
      part += __shfl_xor(part, 8, 16);
      if (l15 == 0) atomicAdd(&vdot[mg], part);
    }
  } else {
    #pragma unroll
    for (int r = 0; r < 8; ++r) {
      int mg = mbase + r;
      Cf[(size_t)mg * N + ng] = acc[r];
    }
  }
}

// ---------------- bf16 WMMA GEMM (NT): C[m][n] = sum_k A[m][k]*B[n][k] ----------------
// 2x2 register tiling per wave (32x32 output tile; 4 interleaved WMMAs hide the D->C
// dependency). Block = 8 waves (4x2) -> 128 rows x 64 cols.
// B slab (64 rows x 32 k = 4 KB) is block-shared: staged through LDS with
// GLOBAL_LOAD_ASYNC_TO_LDS_B128 (ASYNCcnt DMA), double-buffered so the next tile's
// copy overlaps the current tile's WMMAs. A rows are wave-private: direct global
// b128 loads + global_prefetch_b8 ahead of the pipeline.
__global__ __launch_bounds__(256)
void gemm_bf16_kernel(const __bf16* __restrict__ A, const __bf16* __restrict__ Bm,
                      int K, int N, int mode,
                      float* __restrict__ Cf, __bf16* __restrict__ Cbf,
                      const float* __restrict__ rn, const float* __restrict__ cn,
                      const float* __restrict__ ll, const float* __restrict__ ln,
                      const float* __restrict__ Dotf, float* __restrict__ vdot) {
  __shared__ __attribute__((aligned(64))) __bf16 sB[2][64 * 32];   // 2 x 4 KB

  int tid  = threadIdx.x;
  int wv   = tid >> 5;
  int lane = tid & 31;
  int wr = wv >> 1, wc = wv & 1;
  int nBlock = blockIdx.x * 64;           // block col base (64 B-rows staged in LDS)
  int tm0 = blockIdx.y * 128 + wr * 32;   // wave row base (32 rows)
  int tn0 = nBlock + wc * 32;             // wave col base (32 cols)
  int l15 = lane & 15, hi = lane >> 4;

  const __bf16* arow0 = A + (size_t)(tm0 + l15) * K;
  const __bf16* arow1 = A + (size_t)(tm0 + 16 + l15) * K;

  // Per-thread async-copy slot: one b128 per thread covers the 4 KB slab.
  int cpRow = tid >> 2;                   // 0..63  (B row within slab)
  int cpSeg = tid & 3;                    // 0..3   (16-byte segment within 64-B row)
  const __bf16* gRow = Bm + (size_t)(nBlock + cpRow) * K + cpSeg * 8;
  unsigned ldsRow = (unsigned)(uintptr_t)(&sB[0][0]) + cpRow * 64 + cpSeg * 16;

  // Prologue: kick off the first slab into buffer 0.
  {
    const __bf16* g = gRow;
    unsigned l = ldsRow;
    asm volatile("global_load_async_to_lds_b128 %0, %1, off" :: "v"(l), "v"(g) : "memory");
  }

  v8f acc00 = {}, acc01 = {}, acc10 = {}, acc11 = {};
  for (int kt = 0; kt < K; kt += 32) {
    int ibuf = (kt >> 5) & 1;

    // Current slab's DMA done (only our own copy may be outstanding), then sync
    // the whole block so every wave's eighth of the slab is visible.
    asm volatile("s_wait_asynccnt 0x0" ::: "memory");
    __syncthreads();

    // Overlap: start the next slab's DMA into the other buffer. Safe: that buffer
    // was last read in iteration kt-32, and every wave's DS reads completed before
    // it signalled the barrier above.
    if (kt + 32 < K) {
      const __bf16* g = gRow + kt + 32;
      unsigned l = ldsRow + (unsigned)((ibuf ^ 1) * (64 * 32 * 2));
      asm volatile("global_load_async_to_lds_b128 %0, %1, off" :: "v"(l), "v"(g) : "memory");
    }
    if (kt + 256 < K) {   // uniform: pull A ahead toward L0/L2
      __builtin_prefetch(arow0 + kt + 256, 0, 1);
      __builtin_prefetch(arow1 + kt + 256, 0, 1);
    }

    ABf a0, a1, b0, b1;
    // A 16x32 bf16 fragment: lane-half hi picks K-subgroup; two contiguous 16-B chunks
    a0.h[0] = *(const v8bf*)(arow0 + kt + 8 * hi);
    a0.h[1] = *(const v8bf*)(arow0 + kt + 16 + 8 * hi);
    a1.h[0] = *(const v8bf*)(arow1 + kt + 8 * hi);
    a1.h[1] = *(const v8bf*)(arow1 + kt + 16 + 8 * hi);
    // B 32x16 bf16 fragment: same K layout as A (two 16-B chunks per lane), from LDS
    b0.h[0] = *(const v8bf*)(&sB[ibuf][(wc * 32 + l15) * 32 + 8 * hi]);
    b0.h[1] = *(const v8bf*)(&sB[ibuf][(wc * 32 + l15) * 32 + 16 + 8 * hi]);
    b1.h[0] = *(const v8bf*)(&sB[ibuf][(wc * 32 + 16 + l15) * 32 + 8 * hi]);
    b1.h[1] = *(const v8bf*)(&sB[ibuf][(wc * 32 + 16 + l15) * 32 + 16 + 8 * hi]);

    acc00 = __builtin_amdgcn_wmma_f32_16x16x32_bf16(false, a0.v, false, b0.v, (short)0, acc00, false, false);
    acc01 = __builtin_amdgcn_wmma_f32_16x16x32_bf16(false, a0.v, false, b1.v, (short)0, acc01, false, false);
    acc10 = __builtin_amdgcn_wmma_f32_16x16x32_bf16(false, a1.v, false, b0.v, (short)0, acc10, false, false);
    acc11 = __builtin_amdgcn_wmma_f32_16x16x32_bf16(false, a1.v, false, b1.v, (short)0, acc11, false, false);
  }

  float sc = 0.f, noise = 0.f;
  if (mode <= 1) {
    float ls2 = expf(2.0f * ll[0]);
    sc    = -0.5f / ls2;
    noise = expf(ln[0]);
  }
  int mb0 = tm0 + 8 * hi, mb1 = tm0 + 16 + 8 * hi;
  int ng0 = tn0 + l15,    ng1 = tn0 + 16 + l15;
  epilogue_bf16(acc00, mb0, ng0, N, mode, Cf, Cbf, rn, cn, sc, noise, Dotf, vdot, l15);
  epilogue_bf16(acc01, mb0, ng1, N, mode, Cf, Cbf, rn, cn, sc, noise, Dotf, vdot, l15);
  epilogue_bf16(acc10, mb1, ng0, N, mode, Cf, Cbf, rn, cn, sc, noise, Dotf, vdot, l15);
  epilogue_bf16(acc11, mb1, ng1, N, mode, Cf, Cbf, rn, cn, sc, noise, Dotf, vdot, l15);
}

// ---------------- f32 WMMA GEMM (NN, square NxN): C = A*B ----------------
// Newton-Schulz iterations, full f32 via V_WMMA_F32_16X16X4_F32, 2x2 wave tiling.
// mode 0: C = A*B            (T = Kii * X)
// mode 1: C = 2*Xold - A*B   (Xnew = 2X - X*(Kii*X))
__global__ __launch_bounds__(256)
void gemm_f32_kernel(const float* __restrict__ A, const float* __restrict__ Bm,
                     float* __restrict__ C, const float* __restrict__ Xold,
                     int N, int mode) {
  int wv   = threadIdx.x >> 5;
  int lane = threadIdx.x & 31;
  int wr = wv >> 1, wc = wv & 1;
  int tm0 = blockIdx.y * 128 + wr * 32;
  int tn0 = blockIdx.x * 64 + wc * 32;
  int l15 = lane & 15, hi = lane >> 4;

  const float* arow0 = A + (size_t)(tm0 + l15) * N;
  const float* arow1 = A + (size_t)(tm0 + 16 + l15) * N;

  v8f acc00 = {}, acc01 = {}, acc10 = {}, acc11 = {};
  #pragma unroll 8
  for (int kt = 0; kt < N; kt += 4) {
    int kb = kt + 2 * hi;
    if ((kt & 63) == 0 && kt + 64 < N) {
      __builtin_prefetch(arow0 + kt + 64, 0, 1);
      __builtin_prefetch(arow1 + kt + 64, 0, 1);
    }
    // A 16x4 f32 fragment: lane m, elems K = 2*hi + {0,1} (contiguous pair)
    v2f a0 = *(const v2f*)(arow0 + kb);
    v2f a1 = *(const v2f*)(arow1 + kb);
    // B 4x16 f32 fragment: lane n, K = 2*hi + {0,1} (stride-N rows)
    const float* bp0 = Bm + (size_t)kb * N + tn0 + l15;
    const float* bp1 = bp0 + 16;
    v2f b0, b1;
    b0.x = bp0[0]; b0.y = bp0[N];
    b1.x = bp1[0]; b1.y = bp1[N];
    acc00 = __builtin_amdgcn_wmma_f32_16x16x4_f32(false, a0, false, b0, (short)0, acc00, false, false);
    acc01 = __builtin_amdgcn_wmma_f32_16x16x4_f32(false, a0, false, b1, (short)0, acc01, false, false);
    acc10 = __builtin_amdgcn_wmma_f32_16x16x4_f32(false, a1, false, b0, (short)0, acc10, false, false);
    acc11 = __builtin_amdgcn_wmma_f32_16x16x4_f32(false, a1, false, b1, (short)0, acc11, false, false);
  }

  int mb[2] = { tm0 + 8 * hi, tm0 + 16 + 8 * hi };
  int ng[2] = { tn0 + l15,    tn0 + 16 + l15 };
  v8f accs[2][2] = { { acc00, acc01 }, { acc10, acc11 } };
  #pragma unroll
  for (int i = 0; i < 2; ++i) {
    #pragma unroll
    for (int j = 0; j < 2; ++j) {
      #pragma unroll
      for (int r = 0; r < 8; ++r) {
        size_t idx = (size_t)(mb[i] + r) * N + ng[j];
        if (mode == 0) C[idx] = accs[i][j][r];
        else           C[idx] = 2.0f * Xold[idx] - accs[i][j][r];
      }
    }
  }
}

// ---------------- host-side launcher ----------------

extern "C" void kernel_launch(void* const* d_in, const int* in_sizes, int n_in,
                              void* d_out, int out_size, void* d_ws, size_t ws_size,
                              hipStream_t stream) {
  (void)in_sizes; (void)n_in; (void)out_size; (void)ws_size;

  const float* x  = (const float*)d_in[0];   // (B,S,D)
  const float* ip = (const float*)d_in[1];   // (M,D)
  const float* ll = (const float*)d_in[2];   // log_lengthscale (1)
  const float* ln = (const float*)d_in[3];   // log_noise (1)
  const float* W  = (const float*)d_in[4];   // (O,M)

  // bump allocator over workspace (total ~192 MB)
  char* ws = (char*)d_ws;
  size_t off = 0;
  auto alloc = [&](size_t bytes) -> char* {
    char* p = ws + off;
    off = (off + bytes + 255) & ~(size_t)255;
    return p;
  };
  float*    xnorm    = (float*)alloc((size_t)R_ * 4);
  float*    inorm    = (float*)alloc((size_t)M_ * 4);
  float*    vdot     = (float*)alloc((size_t)R_ * 4);
  unsigned* normbits = (unsigned*)alloc(256);
  __bf16*   x_bf     = (__bf16*)alloc((size_t)R_ * D_ * 2);
  __bf16*   ip_bf    = (__bf16*)alloc((size_t)M_ * D_ * 2);
  __bf16*   w_bf     = (__bf16*)alloc((size_t)O_ * M_ * 2);
  float*    Kxi_f    = (float*)alloc((size_t)R_ * M_ * 4);
  __bf16*   Kxi_bf   = (__bf16*)alloc((size_t)R_ * M_ * 2);
  float*    Kii      = (float*)alloc((size_t)M_ * M_ * 4);
  float*    Xb0      = (float*)alloc((size_t)M_ * M_ * 4);
  float*    Xb1      = (float*)alloc((size_t)M_ * M_ * 4);
  float*    Tb       = (float*)alloc((size_t)M_ * M_ * 4);
  __bf16*   Kinv_bf  = (__bf16*)alloc((size_t)M_ * M_ * 2);
  __bf16*   A_bf     = (__bf16*)alloc((size_t)R_ * M_ * 2);

  float* mean_out = (float*)d_out;                          // first R_*O_ floats
  float* var_out  = (float*)d_out + (size_t)R_ * O_;        // second R_*O_ floats

  // 1) row squared norms + bf16 conversions
  rowsumsq_kernel<<<R_, 256, 0, stream>>>(x, xnorm, D_);
  rowsumsq_kernel<<<M_, 256, 0, stream>>>(ip, inorm, D_);
  cvt_bf16_kernel<<<(R_ * D_) / 256, 256, 0, stream>>>(x, x_bf, R_ * D_);
  cvt_bf16_kernel<<<(M_ * D_) / 256, 256, 0, stream>>>(ip, ip_bf, M_ * D_);
  cvt_bf16_kernel<<<(O_ * M_) / 256, 256, 0, stream>>>(W, w_bf, O_ * M_);
  zero_kernel<<<64, 256, 0, stream>>>(vdot, R_);
  zero_kernel<<<1, 64, 0, stream>>>((float*)normbits, 64);

  // 2) K_xi = rbf(x, ip): (16384x1024)*(1024x1024)^T, RBF epilogue, f32 + bf16 out
  gemm_bf16_kernel<<<dim3(M_ / 64, R_ / 128), 256, 0, stream>>>(
      x_bf, ip_bf, D_, M_, 0, Kxi_f, Kxi_bf, xnorm, inorm, ll, ln, nullptr, nullptr);

  // 3) K_ii_reg = rbf(ip, ip) + exp(log_noise)*I
  gemm_bf16_kernel<<<dim3(M_ / 64, M_ / 128), 256, 0, stream>>>(
      ip_bf, ip_bf, D_, M_, 1, Kii, nullptr, inorm, inorm, ll, ln, nullptr, nullptr);

  // 4) Newton-Schulz inverse of K_ii_reg (SPD), all-f32 WMMA
  rowabsmax_kernel<<<M_, 256, 0, stream>>>(Kii, normbits);
  scaleinit_kernel<<<4096, 256, 0, stream>>>(Kii, Xb0, normbits, M_ * M_);
  float* cur = Xb0;
  float* nxt = Xb1;
  for (int it = 0; it < NS_ITERS; ++it) {
    gemm_f32_kernel<<<dim3(M_ / 64, M_ / 128), 256, 0, stream>>>(Kii, cur, Tb, nullptr, M_, 0);
    gemm_f32_kernel<<<dim3(M_ / 64, M_ / 128), 256, 0, stream>>>(cur, Tb, nxt, cur, M_, 1);
    float* t = cur; cur = nxt; nxt = t;
  }
  cvt_bf16_kernel<<<(M_ * M_) / 256, 256, 0, stream>>>(cur, Kinv_bf, M_ * M_);

  // 5) A = K_xi * K_ii_inv (inverse symmetric -> rows usable as NT B operand);
  //    fused: vdot[row] += sum_n A[row][n]*K_xi[row][n]
  gemm_bf16_kernel<<<dim3(M_ / 64, R_ / 128), 256, 0, stream>>>(
      Kxi_bf, Kinv_bf, M_, M_, 2, nullptr, A_bf, nullptr, nullptr, nullptr, nullptr, Kxi_f, vdot);

  // 6) mean = A * W^T  -> straight into d_out
  gemm_bf16_kernel<<<dim3(O_ / 64, R_ / 128), 256, 0, stream>>>(
      A_bf, w_bf, M_, O_, 3, mean_out, nullptr, nullptr, nullptr, nullptr, nullptr, nullptr, nullptr);

  // 7) var_out = softplus(1 - vdot) broadcast over O
  varout_kernel<<<8192, 256, 0, stream>>>(vdot, var_out);
}
